// SelfAttention_31679678775564
// MI455X (gfx1250) — matrix operand order repaired and checked by reading, not compiled, not run on hardware
//
#include <hip/hip_runtime.h>
#include <hip/hip_bf16.h>

// ---------------------------------------------------------------------------
// Self-attention (B=4, N=2048, D=1024), fp32 in/out, bf16 WMMA internally.
//   WqT..WmT = transpose(W*) in bf16 (done once; makes every GEMM A@B^T form)
//   q,k = x@W* + b* (bf16) ; v is written transposed (vT) from the epilogue
//   s   = q @ k^T -> f32 ; aT = softmax over axis=1, stored transposed (bf16)
//   t   = aT @ v  (B = vT, n-major) ; out = relu(t@Wm+bm) -> f32
// All GEMMs: block tile 128x128, wave tile 32x64, K-step 32, double-buffered
// LDS, async global->LDS staging (ASYNCcnt) on gfx1250.
// ---------------------------------------------------------------------------

typedef __attribute__((ext_vector_type(16))) __bf16 v16bf;
typedef __attribute__((ext_vector_type(8)))  __bf16 v8bf;
typedef __attribute__((ext_vector_type(8)))  float  v8f;
typedef int gvi4 __attribute__((vector_size(16)));  // native vector, not HIP int4

#define EMB 1024
#define SEQ 2048
#define BATCH 4
#define LROW 40              // padded LDS row stride (bf16 units) = 80B
#define LTILE (128 * LROW)   // one 128x32 tile in LDS (bf16 units)

#if defined(__HIP_DEVICE_COMPILE__) && defined(__has_builtin)
#if __has_builtin(__builtin_amdgcn_global_load_async_to_lds_b128) && \
    __has_builtin(__builtin_amdgcn_s_wait_asynccnt)
#define USE_ASYNC_LDS 1
#endif
#endif

// Copy 16 bf16 (32B) global -> LDS as two 16B chunks (chunks 16B apart on
// both sides, so the async op's shared immediate offset works for chunk 2).
__device__ __forceinline__ void stage16(const __bf16* g, __bf16* l) {
#ifdef USE_ASYNC_LDS
  gvi4* gp = (gvi4*)(uintptr_t)g;
  gvi4* lp = (gvi4*)(uintptr_t)l;
  __builtin_amdgcn_global_load_async_to_lds_b128(gp, lp, 0, 0);
  __builtin_amdgcn_global_load_async_to_lds_b128(gp, lp, 16, 0);
#else
  *(uint4*)l       = *(const uint4*)g;
  *(uint4*)(l + 8) = *(const uint4*)(g + 8);
#endif
}

__device__ __forceinline__ void wait_stage() {
#ifdef USE_ASYNC_LDS
  __builtin_amdgcn_s_wait_asynccnt(0);
#endif
}

// ------------------------------- fp32 -> bf16 ------------------------------
__global__ void cvt_f32_bf16(const float* __restrict__ in,
                             __bf16* __restrict__ out, size_t n) {
  size_t i = (size_t)blockIdx.x * blockDim.x + threadIdx.x;
  size_t stride = (size_t)gridDim.x * blockDim.x;
  for (; i < n; i += stride) out[i] = (__bf16)in[i];
}

// ------------------- fp32 [R][C] -> bf16 transposed [C][R] -----------------
__global__ __launch_bounds__(256) void cvt_t_f32_bf16(
    const float* __restrict__ in, __bf16* __restrict__ out, int R, int C) {
  __shared__ float t[32][33];
  const int c0 = blockIdx.x * 32, r0 = blockIdx.y * 32;
  const int tx = threadIdx.x & 31, ty = threadIdx.x >> 5;  // 32 x 8
#pragma unroll
  for (int rr = ty; rr < 32; rr += 8)
    t[rr][tx] = in[(size_t)(r0 + rr) * C + c0 + tx];
  __syncthreads();
#pragma unroll
  for (int rr = ty; rr < 32; rr += 8)
    out[(size_t)(c0 + rr) * R + r0 + tx] = (__bf16)t[tx][rr];
}

// ------------------------------ WMMA GEMM ----------------------------------
// C = act(A[M,K] @ B^T + bias), A row-major (lda=K), B given n-major [N][ldb].
// OUT_MODE: 0 = f32 C[M,N] ; 1 = bf16 C[M,N] ; 2 = bf16 transposed C[N][ldcT].
template <bool RELU, int OUT_MODE>
__global__ __launch_bounds__(256) void gemm_nt_wmma(
    const __bf16* __restrict__ A, const __bf16* __restrict__ B,
    const float* __restrict__ bias, void* __restrict__ Cp, int M, int N, int K,
    int ldb, long long sA, long long sB, long long sC, int ldcT) {
  __shared__ __align__(16) __bf16 lA[2 * LTILE];
  __shared__ __align__(16) __bf16 lB[2 * LTILE];

  const int tid = threadIdx.x;
  const int lane = tid & 31, wave = tid >> 5;
  const int wm = wave >> 1, wn = wave & 1;       // 4x2 wave grid
  const int m0 = blockIdx.y * 128, n0 = blockIdx.x * 128;
  const int l15 = lane & 15, lh = lane >> 4;

  A += (size_t)blockIdx.z * sA;
  B += (size_t)blockIdx.z * sB;
  const size_t cbase = (size_t)blockIdx.z * sC;

  // staging assignment: thread moves one 16-bf16 chunk of each 128x32 tile
  const int srow = tid >> 1, sch = (tid & 1) * 16;
  const __bf16* gA = A + (size_t)(m0 + srow) * K + sch;
  const __bf16* gB = B + (size_t)(n0 + srow) * ldb + sch;
  __bf16* dA = &lA[srow * LROW + sch];
  __bf16* dB = &lB[srow * LROW + sch];

  v8f acc[2][4] = {};

#define STAGE_AB(buf, kk)                       \
  do {                                          \
    stage16(gA + (kk), dA + (buf) * LTILE);     \
    stage16(gB + (kk), dB + (buf) * LTILE);     \
  } while (0)

  // Load ALL fragments into distinct registers first, then a clean burst of 8
  // WMMAs (avoids per-pair s_wait_dscnt full-drain stalls on the matrix pipe).
#define MMA_STEP(buf)                                                         \
  do {                                                                        \
    const __bf16* sAp = &lA[(buf) * LTILE];                                   \
    const __bf16* sBp = &lB[(buf) * LTILE];                                   \
    v16bf af[2], bf[4];                                                       \
    _Pragma("unroll") for (int i = 0; i < 2; ++i) {                           \
      const __bf16* p = sAp + (wm * 32 + i * 16 + l15) * LROW + lh * 8;       \
      v8bf lo = *(const v8bf*)p;                                              \
      v8bf hi = *(const v8bf*)(p + 16);                                       \
      af[i] = __builtin_shufflevector(lo, hi, 0, 1, 2, 3, 4, 5, 6, 7, 8, 9,   \
                                      10, 11, 12, 13, 14, 15);                \
    }                                                                         \
    _Pragma("unroll") for (int t = 0; t < 4; ++t) {                           \
      const __bf16* p = sBp + (wn * 64 + t * 16 + l15) * LROW + lh * 8;       \
      v8bf lo = *(const v8bf*)p;                                              \
      v8bf hi = *(const v8bf*)(p + 16);                                       \
      bf[t] = __builtin_shufflevector(lo, hi, 0, 1, 2, 3, 4, 5, 6, 7, 8, 9,   \
                                      10, 11, 12, 13, 14, 15);                \
    }                                                                         \
    _Pragma("unroll") for (int t = 0; t < 4; ++t) {                           \
      acc[0][t] = __builtin_amdgcn_wmma_f32_16x16x32_bf16(                    \
          false, af[0], false, bf[t], (short)0, acc[0][t], false, false);     \
      acc[1][t] = __builtin_amdgcn_wmma_f32_16x16x32_bf16(                    \
          false, af[1], false, bf[t], (short)0, acc[1][t], false, false);     \
    }                                                                         \
  } while (0)

  STAGE_AB(0, 0);
  wait_stage();
  __syncthreads();

  for (int k0 = 0; k0 < K; k0 += 64) {  // K % 64 == 0 for all call sites
    STAGE_AB(1, k0 + 32);
#ifndef USE_ASYNC_LDS
    if (k0 + 64 < K) __builtin_prefetch(gA + k0 + 64, 0, 0);
#endif
    MMA_STEP(0);
    wait_stage();
    __syncthreads();
    if (k0 + 64 < K) STAGE_AB(0, k0 + 64);
    MMA_STEP(1);
    wait_stage();
    __syncthreads();
  }

  // ---- epilogue: C/D layout -> VGPR r holds row M = r + 8*(lane>=16) ----
#pragma unroll
  for (int i = 0; i < 2; ++i) {
    const int rbase = m0 + wm * 32 + i * 16 + lh * 8;
#pragma unroll
    for (int t = 0; t < 4; ++t) {
      const int cc = n0 + wn * 64 + t * 16 + l15;
      const float bv_ = bias ? bias[cc] : 0.0f;
      if (OUT_MODE == 2) {  // transposed bf16: 8 M-contiguous -> one b128
        v8bf pk;
#pragma unroll
        for (int r = 0; r < 8; ++r) pk[r] = (__bf16)(acc[i][t][r] + bv_);
        *(v8bf*)((__bf16*)Cp + cbase + (size_t)cc * ldcT + rbase) = pk;
      } else {
#pragma unroll
        for (int r = 0; r < 8; ++r) {
          float v = acc[i][t][r] + bv_;
          if (RELU) v = fmaxf(v, 0.0f);
          const size_t idx = cbase + (size_t)(rbase + r) * N + cc;
          if (OUT_MODE == 1) ((__bf16*)Cp)[idx] = (__bf16)v;
          else               ((float*)Cp)[idx] = v;
        }
      }
    }
  }
#undef STAGE_AB
#undef MMA_STEP
}

// -------------------- softmax over axis=1, emit alpha^T --------------------
__global__ __launch_bounds__(256) void softmax_ax1_T(
    const float* __restrict__ scores, __bf16* __restrict__ alphaT) {
  const int b = blockIdx.y;
  const int j = blockIdx.x * 256 + threadIdx.x;
  const float* S = scores + (size_t)b * SEQ * SEQ;
  __bf16* AT = alphaT + (size_t)b * SEQ * SEQ + (size_t)j * SEQ;

  float m = -3.402823466e38f;
  for (int i = 0; i < SEQ; ++i) m = fmaxf(m, S[(size_t)i * SEQ + j]);
  float sum = 0.0f;
  for (int i = 0; i < SEQ; ++i) sum += __expf(S[(size_t)i * SEQ + j] - m);
  const float inv = 1.0f / sum;
  for (int i = 0; i < SEQ; ++i)
    AT[i] = (__bf16)(__expf(S[(size_t)i * SEQ + j] - m) * inv);
}

// ---------------------------------------------------------------------------
extern "C" void kernel_launch(void* const* d_in, const int* in_sizes, int n_in,
                              void* d_out, int out_size, void* d_ws,
                              size_t ws_size, hipStream_t stream) {
  const float* x  = (const float*)d_in[0];
  const float* Wq = (const float*)d_in[1];
  const float* bq = (const float*)d_in[2];
  const float* Wk = (const float*)d_in[3];
  const float* bk = (const float*)d_in[4];
  const float* Wv = (const float*)d_in[5];
  const float* bv = (const float*)d_in[6];
  const float* Wm = (const float*)d_in[7];
  const float* bm = (const float*)d_in[8];
  float* out = (float*)d_out;

  const int TOK = BATCH * SEQ;  // 8192
  const size_t XE = (size_t)TOK * EMB;

  char* ws = (char*)d_ws;
  __bf16* xb  = (__bf16*)(ws);                    // 16 MB
  __bf16* qb  = (__bf16*)(ws + (16ull << 20));    // 16 MB
  __bf16* kb  = (__bf16*)(ws + (32ull << 20));    // 16 MB
  __bf16* vT  = (__bf16*)(ws + (48ull << 20));    // 16 MB  [EMB][TOK]
  __bf16* tb  = (__bf16*)(ws + (64ull << 20));    // 16 MB
  __bf16* WqT = (__bf16*)(ws + (80ull << 20));    // 2 MB each, [out][in]
  __bf16* WkT = (__bf16*)(ws + (82ull << 20));
  __bf16* WvT = (__bf16*)(ws + (84ull << 20));
  __bf16* WmT = (__bf16*)(ws + (86ull << 20));
  float*  sc  = (float*) (ws + (88ull << 20));    // 64 MB scores
  __bf16* aT  = (__bf16*)(ws + (152ull << 20));   // 32 MB alpha^T

  // 1) conversions (weights transposed so every GEMM is A @ B^T)
  cvt_f32_bf16<<<4096, 256, 0, stream>>>(x, xb, XE);
  {
    dim3 g(EMB / 32, EMB / 32);
    cvt_t_f32_bf16<<<g, 256, 0, stream>>>(Wq, WqT, EMB, EMB);
    cvt_t_f32_bf16<<<g, 256, 0, stream>>>(Wk, WkT, EMB, EMB);
    cvt_t_f32_bf16<<<g, 256, 0, stream>>>(Wv, WvT, EMB, EMB);
    cvt_t_f32_bf16<<<g, 256, 0, stream>>>(Wm, WmT, EMB, EMB);
  }

  // 2) projections [8192x1024]@[1024x1024]^T ; v written transposed
  {
    dim3 g(EMB / 128, TOK / 128, 1);
    gemm_nt_wmma<false, 1><<<g, 256, 0, stream>>>(
        xb, WqT, bq, qb, TOK, EMB, EMB, EMB, 0, 0, 0, 0);
    gemm_nt_wmma<false, 1><<<g, 256, 0, stream>>>(
        xb, WkT, bk, kb, TOK, EMB, EMB, EMB, 0, 0, 0, 0);
    gemm_nt_wmma<false, 2><<<g, 256, 0, stream>>>(
        xb, WvT, bv, vT, TOK, EMB, EMB, EMB, 0, 0, 0, TOK);
  }

  // 3) scores = q @ k^T per batch -> f32
  {
    dim3 g(SEQ / 128, SEQ / 128, BATCH);
    gemm_nt_wmma<false, 0><<<g, 256, 0, stream>>>(
        qb, kb, nullptr, sc, SEQ, SEQ, EMB, EMB,
        (long long)SEQ * EMB, (long long)SEQ * EMB, (long long)SEQ * SEQ, 0);
  }

  // 4) softmax over axis=1 -> alpha^T (bf16)
  {
    dim3 g(SEQ / 256, BATCH);
    softmax_ax1_T<<<g, 256, 0, stream>>>(sc, aT);
  }

  // 5) t = alpha^T @ v per batch (B = vT slice, n-major, ldb = TOK)
  {
    dim3 g(EMB / 128, SEQ / 128, BATCH);
    gemm_nt_wmma<false, 1><<<g, 256, 0, stream>>>(
        aT, vT, nullptr, tb, SEQ, EMB, SEQ, TOK,
        (long long)SEQ * SEQ, (long long)SEQ, (long long)SEQ * EMB, 0);
  }

  // 6) out = relu(t @ Wm + bm) -> f32
  {
    dim3 g(EMB / 128, TOK / 128, 1);
    gemm_nt_wmma<true, 0><<<g, 256, 0, stream>>>(
        tb, WmT, bm, out, TOK, EMB, EMB, EMB, 0, 0, 0, 0);
  }
}
